// TransformerBlock_6863357739241
// MI455X (gfx1250) — compile-verified, matrix operand order checked by reading, not run on hardware
//
#include <hip/hip_runtime.h>

#define DEVI __device__ __forceinline__

typedef __attribute__((ext_vector_type(16))) __bf16 v16bf;
typedef __attribute__((ext_vector_type(8)))  float  v8f;
typedef __attribute__((ext_vector_type(4)))  unsigned int u32x4;

union FragU { v16bf v; u32x4 u[2]; };

constexpr int Dm = 1024, Hh = 16, FF = 4096, NE = 8;
constexpr int Bb = 4, Ss = 1024, Tt = Bb * Ss;      // T = 4096 tokens
constexpr float LNEPS = 1e-5f;
constexpr int MOE_XLD = 1032;   // 1024 + 8 pad (keeps 16B alignment, rotates banks)
constexpr int MOE_HLD = 4104;   // 4096 + 8 pad
constexpr int MOE_LDS_BYTES = 16 * MOE_XLD * 2 + 16 * MOE_HLD * 2 + 256; // +slack for tail prefetch

// ---------- helpers ----------

DEVI __bf16 f2b(float f) {
  unsigned u = __builtin_bit_cast(unsigned, f);
  u += 0x7FFFu + ((u >> 16) & 1u);                  // round-to-nearest-even
  unsigned short s = (unsigned short)(u >> 16);
  return __builtin_bit_cast(__bf16, s);
}

// p already points at this lane's first 16-byte K-group:
//   base + (row0 + lane%16)*ld + (lane>=16 ? 8 : 0) [+ constant offsets at call site]
// lanes 0-15 hold K {0..7, 16..23}; lanes 16-31 hold K {8..15, 24..31}.
DEVI v16bf ldfrag(const __bf16* p) {
  FragU f;
  f.u[0] = *(const u32x4*)(p);
  f.u[1] = *(const u32x4*)(p + 16);
  return f.v;
}

DEVI const __bf16* frag_base(const __bf16* base, int ld, int row0, int lane) {
  return base + (size_t)(row0 + (lane & 15)) * ld + ((lane >> 4) << 3);
}

DEVI v8f wmma_bf16(v16bf a, v16bf b, v8f c) {
  return __builtin_amdgcn_wmma_f32_16x16x32_bf16(false, a, false, b, (short)0, c,
                                                 false, false);
}

// ---------- conversion kernels ----------

__global__ void cvt_kernel(const float* __restrict__ s, __bf16* __restrict__ d, long n) {
  long i = (long)blockIdx.x * blockDim.x + threadIdx.x;
  if (i < n) d[i] = f2b(s[i]);
}

// src: [grid.z][R][C] f32 row-major  ->  dst: [C][R] bf16 (transposed)
__global__ void cvtT_kernel(const float* __restrict__ src, __bf16* __restrict__ dst,
                            int R, int C) {
  __shared__ float tile[32][33];
  int e = blockIdx.z;
  src += (size_t)e * R * C;
  dst += (size_t)e * R * C;
  int c0 = blockIdx.x * 32, r0 = blockIdx.y * 32;
  for (int i = threadIdx.y; i < 32; i += 8)
    tile[i][threadIdx.x] = src[(size_t)(r0 + i) * C + c0 + threadIdx.x];
  __syncthreads();
  for (int i = threadIdx.y; i < 32; i += 8)
    dst[(size_t)(c0 + i) * R + r0 + threadIdx.x] = f2b(tile[threadIdx.x][i]);
}

__global__ void zero_kernel(float* p, int n) {
  int i = blockIdx.x * blockDim.x + threadIdx.x;
  if (i < n) p[i] = 0.0f;
}

// ---------- generic bf16 WMMA GEMM:  C[M,N] = A[M,K] @ W[N,K]^T + bias ----------
// block = 128 threads (4 waves); each wave computes a 16 x 128 output tile.
// K is a template constant so all 8 B-streams share ONE address register with
// compile-time immediate offsets (j*16*K*2 bytes fits the signed 24-bit IOFFSET).

template <int K>
__global__ void gemm_kernel(const __bf16* __restrict__ A, const __bf16* __restrict__ W,
                            const float* __restrict__ bias, float* outF, __bf16* outB,
                            int N) {
  int lane = threadIdx.x & 31;
  int m0 = (blockIdx.y * 4 + (threadIdx.x >> 5)) * 16;
  int n0 = blockIdx.x * 128;

  const __bf16* pA = frag_base(A, K, m0, lane);
  const __bf16* pB = frag_base(W, K, n0, lane);

  v16bf a0 = ldfrag(pA);
  pA += 32;
  v16bf blo[4], bhi[4];
#pragma unroll
  for (int j = 0; j < 4; ++j) blo[j] = ldfrag(pB + j * 16 * K);
  v8f acc[8] = {};
  for (int kt = 0; kt < K; kt += 32) {
#pragma unroll
    for (int j = 0; j < 4; ++j) bhi[j] = ldfrag(pB + (j + 4) * 16 * K);
    pB += 32;
#pragma unroll
    for (int j = 0; j < 4; ++j) acc[j] = wmma_bf16(a0, blo[j], acc[j]);
    v16bf an = ldfrag(pA);            // dead on final iter (stays inside ws)
    pA += 32;
#pragma unroll
    for (int j = 0; j < 4; ++j) blo[j] = ldfrag(pB + j * 16 * K);
#pragma unroll
    for (int j = 0; j < 4; ++j) acc[j + 4] = wmma_bf16(a0, bhi[j], acc[j + 4]);
    a0 = an;
  }

  int col = lane & 15;
  int rb = (lane & 16) ? 8 : 0;
#pragma unroll
  for (int j = 0; j < 8; ++j) {
    int n = n0 + j * 16 + col;
    float bv = bias[n];
#pragma unroll
    for (int v = 0; v < 8; ++v) {
      float c = acc[j][v] + bv;
      size_t idx = (size_t)(m0 + rb + v) * N + n;
      if (outF) outF[idx] = c;
      if (outB) outB[idx] = f2b(c);
    }
  }
}

// ---------- flash attention ----------
// grid.x = B*H, grid.y = S/128, block = 256 (8 waves, each owns 16 q-rows).
// 1 m-tile per wave keeps live state ~110 VGPRs -> no scratch spills.

__global__ void attn_kernel(const __bf16* __restrict__ qkvb, __bf16* __restrict__ ctxb) {
  constexpr int KLD = 72, VLD = 40, PLD = 40;
  __shared__ __bf16 kt[32 * KLD];       // K block, row-major [s_rel][d]
  __shared__ __bf16 vt[64 * VLD];       // V block, transposed [d][s_rel]
  __shared__ __bf16 pp[8 * 16 * PLD];   // per-wave P tile (C-layout -> A-layout)
  int bh = blockIdx.x, b = bh >> 4, h = bh & 15;
  int tid = threadIdx.x, lane = tid & 31, w = tid >> 5;
  int s0 = blockIdx.y * 128 + w * 16;
  int col = lane & 15, rb = (lane & 16) ? 8 : 0;

  const __bf16* qbase = qkvb + (size_t)b * Ss * (3 * Dm) + h * 64;
  v16bf qf[2];
  {
    const __bf16* pq = frag_base(qbase, 3 * Dm, s0, lane);
#pragma unroll
    for (int ks = 0; ks < 2; ++ks) qf[ks] = ldfrag(pq + ks * 32);
  }

  v8f cacc[4] = {};
  float mrun[8], lrun[8];
#pragma unroll
  for (int v = 0; v < 8; ++v) { mrun[v] = -1e30f; lrun[v] = 0.0f; }
  __bf16* pw = pp + w * 16 * PLD;

  for (int kb = 0; kb < Ss; kb += 32) {
    __syncthreads();
    for (int i = tid; i < 32 * 64; i += 256) {
      int d = i & 63, sr = i >> 6;
      size_t tok = (size_t)(b * Ss + kb + sr) * (3 * Dm) + h * 64 + d;
      kt[sr * KLD + d] = qkvb[tok + Dm];        // K
      vt[d * VLD + sr] = qkvb[tok + 2 * Dm];    // V transposed
    }
    // prefetch next block's K/V rows into cache while this block computes
    if (kb + 32 < Ss && tid < 64) {
      int sr = tid & 31;
      const __bf16* pf = qkvb + (size_t)(b * Ss + kb + 32 + sr) * (3 * Dm) + h * 64 +
                         ((tid < 32) ? Dm : 2 * Dm);
      __builtin_prefetch(pf, 0, 3);
    }
    __syncthreads();

    // S = Q @ K^T  (16x32 per wave)
    v8f sc[2] = {};
#pragma unroll
    for (int nt = 0; nt < 2; ++nt) {
      const __bf16* pk = frag_base(kt, KLD, nt * 16, lane);
#pragma unroll
      for (int ks = 0; ks < 2; ++ks) {
        v16bf bf = ldfrag(pk + ks * 32);
        sc[nt] = wmma_bf16(qf[ks], bf, sc[nt]);
      }
    }

    // online softmax; row m = v + 8*(lane>=16); 16-lane shfl reductions
#pragma unroll
    for (int v = 0; v < 8; ++v) {
      float x0 = sc[0][v] * 0.125f;   // 1/sqrt(64)
      float x1 = sc[1][v] * 0.125f;
      float mx = fmaxf(x0, x1);
#pragma unroll
      for (int off = 1; off < 16; off <<= 1)
        mx = fmaxf(mx, __shfl_xor(mx, off, 32));
      float mnew = fmaxf(mrun[v], mx);
      float alpha = __expf(mrun[v] - mnew);
      float p0 = __expf(x0 - mnew);
      float p1 = __expf(x1 - mnew);
      float rs = p0 + p1;
#pragma unroll
      for (int off = 1; off < 16; off <<= 1)
        rs += __shfl_xor(rs, off, 32);
      lrun[v] = lrun[v] * alpha + rs;
      mrun[v] = mnew;
#pragma unroll
      for (int nt = 0; nt < 4; ++nt) cacc[nt][v] *= alpha;
      int r = rb + v;
      pw[r * PLD + col] = f2b(p0);
      pw[r * PLD + 16 + col] = f2b(p1);
    }

    // ctx += P @ V   (k = 32, one WMMA step)
    {
      v16bf pa = ldfrag(frag_base(pw, PLD, 0, lane));
#pragma unroll
      for (int nt = 0; nt < 4; ++nt) {
        v16bf bf = ldfrag(frag_base(vt, VLD, nt * 16, lane));
        cacc[nt] = wmma_bf16(pa, bf, cacc[nt]);
      }
    }
  }

#pragma unroll
  for (int nt = 0; nt < 4; ++nt)
#pragma unroll
    for (int v = 0; v < 8; ++v) {
      float c = cacc[nt][v] / lrun[v];
      int srow = s0 + rb + v;
      int d = nt * 16 + col;
      ctxb[(size_t)(b * Ss + srow) * Dm + h * 64 + d] = f2b(c);
    }
}

// ---------- LayerNorm: out = LN(a + r) * g + b ; optional bf16 copy ----------

__global__ void ln_kernel(const float* __restrict__ a, const float* __restrict__ r,
                          const float* __restrict__ g, const float* __restrict__ be,
                          float* __restrict__ outF, __bf16* outB) {
  __shared__ float red[256];
  int row = blockIdx.x, tid = threadIdx.x;
  const float* pa = a + (size_t)row * Dm;
  const float* pr = r + (size_t)row * Dm;
  float x[4];
  float s = 0.0f;
#pragma unroll
  for (int i = 0; i < 4; ++i) {
    x[i] = pa[tid + 256 * i] + pr[tid + 256 * i];
    s += x[i];
  }
  red[tid] = s; __syncthreads();
  for (int st = 128; st > 0; st >>= 1) { if (tid < st) red[tid] += red[tid + st]; __syncthreads(); }
  float mean = red[0] / (float)Dm; __syncthreads();
  float q = 0.0f;
#pragma unroll
  for (int i = 0; i < 4; ++i) { float d = x[i] - mean; q += d * d; }
  red[tid] = q; __syncthreads();
  for (int st = 128; st > 0; st >>= 1) { if (tid < st) red[tid] += red[tid + st]; __syncthreads(); }
  float inv = rsqrtf(red[0] / (float)Dm + LNEPS);
#pragma unroll
  for (int i = 0; i < 4; ++i) {
    int d = tid + 256 * i;
    float o = (x[i] - mean) * inv * g[d] + be[d];
    outF[(size_t)row * Dm + d] = o;
    if (outB) outB[(size_t)row * Dm + d] = f2b(o);
  }
}

// ---------- router: softmax over E=8, top-2, renorm, aux partial sums ----------
// one wave per token; block = 256 (8 tokens)

__global__ void router_kernel(const float* __restrict__ x1, const float* __restrict__ gw,
                              float* __restrict__ gates, float* __restrict__ auxb) {
  int w = threadIdx.x >> 5, lane = threadIdx.x & 31;
  int t = blockIdx.x * 8 + w;
  const float* xr = x1 + (size_t)t * Dm;
  float acc[NE] = {};
  for (int i = 0; i < Dm / 32; ++i) {
    float xv = xr[lane + 32 * i];
#pragma unroll
    for (int e = 0; e < NE; ++e) acc[e] += xv * gw[e * Dm + lane + 32 * i];
  }
#pragma unroll
  for (int e = 0; e < NE; ++e)
#pragma unroll
    for (int off = 16; off > 0; off >>= 1) acc[e] += __shfl_xor(acc[e], off, 32);
  float mx = acc[0];
#pragma unroll
  for (int e = 1; e < NE; ++e) mx = fmaxf(mx, acc[e]);
  float pe[NE], sum = 0.0f;
#pragma unroll
  for (int e = 0; e < NE; ++e) { pe[e] = __expf(acc[e] - mx); sum += pe[e]; }
#pragma unroll
  for (int e = 0; e < NE; ++e) pe[e] /= sum;
  int i1 = 0; float v1 = pe[0];
#pragma unroll
  for (int e = 1; e < NE; ++e) if (pe[e] > v1) { v1 = pe[e]; i1 = e; }
  int i2 = (i1 == 0) ? 1 : 0; float v2 = pe[i2];
#pragma unroll
  for (int e = 0; e < NE; ++e)
    if (e != i1 && pe[e] > v2) { v2 = pe[e]; i2 = e; }
  float z = v1 + v2;
  if (lane == 0) {
#pragma unroll
    for (int e = 0; e < NE; ++e)
      gates[(size_t)t * NE + e] = (e == i1) ? v1 / z : ((e == i2) ? v2 / z : 0.0f);
#pragma unroll
    for (int e = 0; e < NE; ++e) atomicAdd(&auxb[e], pe[e]);   // probs sums
    atomicAdd(&auxb[8 + i1], 1.0f);                            // dispatch counts
    atomicAdd(&auxb[8 + i2], 1.0f);
  }
}

__global__ void aux_kernel(const float* __restrict__ auxb, float* __restrict__ out) {
  if (threadIdx.x == 0) {
    float a = 0.0f;
    for (int e = 0; e < NE; ++e)
      a += (auxb[8 + e] / (float)Tt) * (auxb[e] / (float)Tt);
    out[0] = ((float)NE / 2.0f) * a;  // (E/K) * sum(frac * probs_mean)
  }
}

// ---------- fused MoE: per 16-token tile, loop experts, gemm1->relu->gemm2 ----------
// block = 256 (8 waves). Dynamic LDS: x-tile [16][1032] bf16 + h-tile [16][4104] bf16.
// Both GEMMs software-pipelined; all B streams share one advancing address with
// compile-time immediate offsets.

__global__ void moe_kernel(const __bf16* __restrict__ x1b, const __bf16* __restrict__ w1t,
                           const __bf16* __restrict__ w2t, const float* __restrict__ b1,
                           const float* __restrict__ b2, const float* __restrict__ gates,
                           float* __restrict__ moe) {
  extern __shared__ char smem[];
  __bf16* xa = (__bf16*)smem;                                   // [16][MOE_XLD]
  __bf16* hh = (__bf16*)(smem + 16 * MOE_XLD * sizeof(__bf16)); // [16][MOE_HLD]
  int t0 = blockIdx.x * 16;
  int tid = threadIdx.x, lane = tid & 31, w = tid >> 5;
  int col = lane & 15, rb = (lane & 16) ? 8 : 0;

  // stage the 16 x 1024 activation tile once (vectorized 16B copies)
  for (int i = tid; i < 16 * (Dm / 8); i += 256) {
    int r = i >> 7, c = (i & 127) * 8;
    *(u32x4*)(xa + r * MOE_XLD + c) = *(const u32x4*)(x1b + (size_t)(t0 + r) * Dm + c);
  }
  __syncthreads();

  v8f mo[8] = {};
  for (int e = 0; e < NE; ++e) {
    const __bf16* W1e = w1t + (size_t)e * FF * Dm;   // [FF][D]
    const __bf16* W2e = w2t + (size_t)e * Dm * FF;   // [D][FF]

    // gemm1: h = relu(x @ W1 + b1); wave w owns FF cols [w*512, w*512+512)
    for (int nc = 0; nc < 4; ++nc) {
      int nb = w * 512 + nc * 128;
      const __bf16* pA = frag_base(xa, MOE_XLD, 0, lane);
      const __bf16* pB = frag_base(W1e, Dm, nb, lane);
      v16bf a0 = ldfrag(pA);
      pA += 32;
      v16bf blo[4], bhi[4];
#pragma unroll
      for (int j = 0; j < 4; ++j) blo[j] = ldfrag(pB + j * 16 * Dm);
      v8f acc[8] = {};
      for (int ktep = 0; ktep < Dm; ktep += 32) {
#pragma unroll
        for (int j = 0; j < 4; ++j) bhi[j] = ldfrag(pB + (j + 4) * 16 * Dm);
        pB += 32;
#pragma unroll
        for (int j = 0; j < 4; ++j) acc[j] = wmma_bf16(a0, blo[j], acc[j]);
        v16bf an = ldfrag(pA);        // dead on final iter (stays in LDS slack)
        pA += 32;
#pragma unroll
        for (int j = 0; j < 4; ++j) blo[j] = ldfrag(pB + j * 16 * Dm);
#pragma unroll
        for (int j = 0; j < 4; ++j) acc[j + 4] = wmma_bf16(a0, bhi[j], acc[j + 4]);
        a0 = an;
      }
#pragma unroll
      for (int j = 0; j < 8; ++j) {
        int n = nb + j * 16 + col;
        float bv = b1[e * FF + n];
#pragma unroll
        for (int v = 0; v < 8; ++v) {
          float c = fmaxf(acc[j][v] + bv, 0.0f);
          hh[(size_t)(rb + v) * MOE_HLD + n] = f2b(c);
        }
      }
    }
    __syncthreads();

    // gemm2: y = h @ W2 + b2; wave w owns D cols [w*128, w*128+128)
    {
      int nb = w * 128;
      const __bf16* pA = frag_base(hh, MOE_HLD, 0, lane);
      const __bf16* pB = frag_base(W2e, FF, nb, lane);
      v16bf a0 = ldfrag(pA);
      pA += 32;
      v16bf blo[4], bhi[4];
#pragma unroll
      for (int j = 0; j < 4; ++j) blo[j] = ldfrag(pB + j * 16 * FF);
      v8f acc[8] = {};
      for (int ktep = 0; ktep < FF; ktep += 32) {
#pragma unroll
        for (int j = 0; j < 4; ++j) bhi[j] = ldfrag(pB + (j + 4) * 16 * FF);
        pB += 32;
#pragma unroll
        for (int j = 0; j < 4; ++j) acc[j] = wmma_bf16(a0, blo[j], acc[j]);
        v16bf an = ldfrag(pA);
        pA += 32;
#pragma unroll
        for (int j = 0; j < 4; ++j) blo[j] = ldfrag(pB + j * 16 * FF);
#pragma unroll
        for (int j = 0; j < 4; ++j) acc[j + 4] = wmma_bf16(a0, bhi[j], acc[j + 4]);
        a0 = an;
      }
      float gv[8];
#pragma unroll
      for (int v = 0; v < 8; ++v) gv[v] = gates[(size_t)(t0 + rb + v) * NE + e];
#pragma unroll
      for (int j = 0; j < 8; ++j) {
        float bv = b2[e * Dm + nb + j * 16 + col];
#pragma unroll
        for (int v = 0; v < 8; ++v) mo[j][v] += gv[v] * (acc[j][v] + bv);
      }
    }
    __syncthreads();
  }
#pragma unroll
  for (int j = 0; j < 8; ++j)
#pragma unroll
    for (int v = 0; v < 8; ++v)
      moe[(size_t)(t0 + rb + v) * Dm + w * 128 + j * 16 + col] = mo[j][v];
}

// ---------- host-side orchestration ----------

extern "C" void kernel_launch(void* const* d_in, const int* in_sizes, int n_in,
                              void* d_out, int out_size, void* d_ws, size_t ws_size,
                              hipStream_t stream) {
  (void)in_sizes; (void)n_in; (void)out_size; (void)ws_size;
  const float* x    = (const float*)d_in[0];
  const float* Wqkv = (const float*)d_in[1];
  const float* bqkv = (const float*)d_in[2];
  const float* Wo   = (const float*)d_in[3];
  const float* bo   = (const float*)d_in[4];
  const float* g1   = (const float*)d_in[5];
  const float* be1  = (const float*)d_in[6];
  const float* g2   = (const float*)d_in[7];
  const float* be2  = (const float*)d_in[8];
  const float* gw   = (const float*)d_in[9];
  const float* W1   = (const float*)d_in[10];
  const float* b1   = (const float*)d_in[11];
  const float* W2   = (const float*)d_in[12];
  const float* b2   = (const float*)d_in[13];
  float* out = (float*)d_out;

  char* ws = (char*)d_ws;
  size_t off = 0;
  auto carve = [&](size_t bytes) -> char* {
    char* p = ws + off;
    off += (bytes + 255) & ~(size_t)255;
    return p;
  };
  __bf16* xbf  = (__bf16*)carve((size_t)Tt * Dm * 2);
  __bf16* wqkv = (__bf16*)carve((size_t)3 * Dm * Dm * 2);
  __bf16* wout = (__bf16*)carve((size_t)Dm * Dm * 2);
  __bf16* w1t  = (__bf16*)carve((size_t)NE * FF * Dm * 2);
  __bf16* w2t  = (__bf16*)carve((size_t)NE * Dm * FF * 2);
  __bf16* qkvb = (__bf16*)carve((size_t)Tt * 3 * Dm * 2);
  __bf16* ctxb = (__bf16*)carve((size_t)Tt * Dm * 2);
  float*  attn = (float*)carve((size_t)Tt * Dm * 4);
  float*  x1   = (float*)carve((size_t)Tt * Dm * 4);
  __bf16* x1b  = (__bf16*)carve((size_t)Tt * Dm * 2);
  float*  moe  = (float*)carve((size_t)Tt * Dm * 4);
  float*  gts  = (float*)carve((size_t)Tt * NE * 4);
  float*  auxb = (float*)carve(64 + 256);   // final carve keeps tail-prefetch slack

  (void)hipFuncSetAttribute((const void*)moe_kernel,
                            hipFuncAttributeMaxDynamicSharedMemorySize, MOE_LDS_BYTES);

  zero_kernel<<<1, 64, 0, stream>>>(auxb, 16);
  cvt_kernel<<<(Tt * Dm) / 256, 256, 0, stream>>>(x, xbf, (long)Tt * Dm);
  cvt_kernel<<<(3 * Dm * Dm) / 256, 256, 0, stream>>>(Wqkv, wqkv, (long)3 * Dm * Dm);
  cvt_kernel<<<(Dm * Dm) / 256, 256, 0, stream>>>(Wo, wout, (long)Dm * Dm);
  cvtT_kernel<<<dim3(FF / 32, Dm / 32, NE), dim3(32, 8), 0, stream>>>(W1, w1t, Dm, FF);
  cvtT_kernel<<<dim3(Dm / 32, FF / 32, NE), dim3(32, 8), 0, stream>>>(W2, w2t, FF, Dm);

  gemm_kernel<Dm><<<dim3(3 * Dm / 128, Tt / 64), 128, 0, stream>>>(
      xbf, wqkv, bqkv, nullptr, qkvb, 3 * Dm);
  attn_kernel<<<dim3(Bb * Hh, Ss / 128), 256, 0, stream>>>(qkvb, ctxb);
  gemm_kernel<Dm><<<dim3(Dm / 128, Tt / 64), 128, 0, stream>>>(
      ctxb, wout, bo, attn, nullptr, Dm);
  ln_kernel<<<Tt, 256, 0, stream>>>(x, attn, g1, be1, x1, x1b);
  router_kernel<<<Tt / 8, 256, 0, stream>>>(x1, gw, gts, auxb);
  moe_kernel<<<Tt / 16, 256, MOE_LDS_BYTES, stream>>>(x1b, w1t, w2t, b1, b2, gts, moe);
  ln_kernel<<<Tt, 256, 0, stream>>>(x1, moe, g2, be2, out, nullptr);
  aux_kernel<<<1, 32, 0, stream>>>(auxb, out + (size_t)Tt * Dm);
}
